// Seq2SeqLSTM_17463337025828
// MI455X (gfx1250) — matrix operand order repaired and checked by reading, not compiled, not run on hardware
//
#include <hip/hip_runtime.h>
#include <hip/hip_bf16.h>

// Problem constants (from reference)
#define B_    256
#define T_IN  336
#define T_OUT 96
#define F_    32
#define H_    256
#define G_    1024   // 4*H

typedef __attribute__((ext_vector_type(8)))  __bf16 v8bf;
typedef __attribute__((ext_vector_type(16))) __bf16 v16bf;
typedef __attribute__((ext_vector_type(8)))  float  v8f;

// round-to-nearest-even f32 -> bf16 without relying on __bf16 conversions
__device__ __forceinline__ __bf16 f2bf(float f) {
  union { float f; unsigned u; } v; v.f = f;
  unsigned r = v.u + 0x7FFFu + ((v.u >> 16) & 1u);
  unsigned short h = (unsigned short)(r >> 16);
  return __builtin_bit_cast(__bf16, h);
}

// ---- WMMA fragment loaders (CDNA5 16-bit layouts, wave32) ----
// A 16x32 (MxK): lane m = lane&15; half=lane>>4;
//   v16 elems 0..7  = K[kbase + half*8 .. +7]
//   v16 elems 8..15 = K[kbase + 16 + half*8 .. +7]
__device__ __forceinline__ v16bf load_a(const __bf16* A, int lda, int kbase, int lane) {
  const int m = lane & 15, half = lane >> 4;
  const __bf16* p = A + m * lda + kbase + half * 8;
  v8bf lo = *(const v8bf*)p;          // 16B aligned
  v8bf hi = *(const v8bf*)(p + 16);   // 16B aligned
  v16bf r;
#pragma unroll
  for (int i = 0; i < 8; ++i) { r[i] = lo[i]; r[i + 8] = hi[i]; }
  return r;
}

// B 32x16 (KxN) with B[k][n] = W[n][k] (W row-major [4H][K]):
//   lane n = lane&15; half=lane>>4; elems 0..7 = K[kbase + half*8 .. +7], elems 8..15 = K[kbase + 16 + half*8 .. +7]
__device__ __forceinline__ v16bf load_b(const __bf16* W, int ldw, int n, int kbase, int lane) {
  const int half = lane >> 4;
  const __bf16* p = W + n * ldw + kbase + half * 8;
  v8bf lo = *(const v8bf*)p;
  v8bf hi = *(const v8bf*)(p + 16);
  v16bf r;
#pragma unroll
  for (int i = 0; i < 8; ++i) { r[i] = lo[i]; r[i + 8] = hi[i]; }
  return r;
}

// gates[16 x 1024] = bias + A1[16 x K1] @ W1^T + A2[16 x K2] @ W2^T
// 16 waves; wave owns 64 columns (4 tiles of 16); f32 accumulation.
__device__ __forceinline__ void gemm_gates(
    const __bf16* A1, int lda1, const __bf16* W1, int ldw1, int K1,
    const __bf16* A2, int lda2, const __bf16* W2, int ldw2, int K2,
    const float* __restrict__ bias, float* gates, int wave, int lane) {
  const int nib = lane & 15, half = lane >> 4;
  const int colbase = wave * 64;
  v8f acc[4];
#pragma unroll
  for (int ct = 0; ct < 4; ++ct) {
    float b = bias[colbase + ct * 16 + nib];
#pragma unroll
    for (int i = 0; i < 8; ++i) acc[ct][i] = b;
  }
#pragma unroll 1
  for (int kt = 0; kt < K1; kt += 32) {
    v16bf a = load_a(A1, lda1, kt, lane);
#pragma unroll
    for (int ct = 0; ct < 4; ++ct) {
      v16bf b = load_b(W1, ldw1, colbase + ct * 16 + nib, kt, lane);
      acc[ct] = __builtin_amdgcn_wmma_f32_16x16x32_bf16(
          false, a, false, b, (short)0, acc[ct], false, false);
    }
  }
#pragma unroll 1
  for (int kt = 0; kt < K2; kt += 32) {
    v16bf a = load_a(A2, lda2, kt, lane);
#pragma unroll
    for (int ct = 0; ct < 4; ++ct) {
      v16bf b = load_b(W2, ldw2, colbase + ct * 16 + nib, kt, lane);
      acc[ct] = __builtin_amdgcn_wmma_f32_16x16x32_bf16(
          false, a, false, b, (short)0, acc[ct], false, false);
    }
  }
  // C/D layout: VGPR r -> row m = r + half*8, col n = colbase + ct*16 + nib
#pragma unroll
  for (int ct = 0; ct < 4; ++ct)
#pragma unroll
    for (int r = 0; r < 8; ++r)
      gates[(r + half * 8) * G_ + colbase + ct * 16 + nib] = acc[ct][r];
}

// LSTM cell elementwise over 16 x 256 state, 512 threads
__device__ __forceinline__ void lstm_cell(const float* gates, float* c, __bf16* h, int tid) {
#pragma unroll
  for (int it = 0; it < 8; ++it) {
    int idx = it * 512 + tid;       // 0..4095
    int m = idx >> 8, hh = idx & 255;
    const float* gm = gates + m * G_;
    float ig = gm[hh], fg = gm[256 + hh], gg = gm[512 + hh], og = gm[768 + hh];
    float si = 1.f / (1.f + __expf(-ig));
    float sf = 1.f / (1.f + __expf(-fg));
    float so = 1.f / (1.f + __expf(-og));
    float e2 = __expf(2.f * gg);
    float tg = (e2 - 1.f) / (e2 + 1.f);
    float cn = sf * c[idx] + si * tg;
    c[idx] = cn;
    float ec = __expf(2.f * cn);
    float tc = (ec - 1.f) / (ec + 1.f);
    h[idx] = f2bf(so * tc);
  }
}

__global__ void f32_to_bf16_k(const float* __restrict__ s, __bf16* __restrict__ d, int n) {
  int i = blockIdx.x * blockDim.x + threadIdx.x;
  if (i < n) d[i] = f2bf(s[i]);
}

__global__ __launch_bounds__(512)
void lstm_seq2seq_k(const __bf16* __restrict__ xb,
                    const __bf16* __restrict__ eWih0, const __bf16* __restrict__ eWhh0,
                    const float*  __restrict__ eb0,
                    const __bf16* __restrict__ eWih1, const __bf16* __restrict__ eWhh1,
                    const float*  __restrict__ eb1,
                    const __bf16* __restrict__ dWih0, const __bf16* __restrict__ dWhh0,
                    const float*  __restrict__ db0,
                    const __bf16* __restrict__ dWih1, const __bf16* __restrict__ dWhh1,
                    const float*  __restrict__ db1,
                    const __bf16* __restrict__ dWout, const float* __restrict__ dbout,
                    float* __restrict__ out) {
  extern __shared__ char smem[];
  float*  gates = (float*)smem;                          // 64 KB
  float*  c0    = (float*)(smem + 65536);                // 16 KB
  float*  c1    = (float*)(smem + 65536 + 16384);        // 16 KB
  __bf16* h0    = (__bf16*)(smem + 65536 + 32768);       // 8 KB
  __bf16* h1    = (__bf16*)(smem + 65536 + 32768 + 8192);// 8 KB
  __bf16* yb    = (__bf16*)(smem + 65536 + 32768 + 16384); // 1 KB (16x32)

  const int tid = threadIdx.x, lane = tid & 31, wave = tid >> 5;
  const int b0 = blockIdx.x * 16;   // batch tile of 16 rows

  const __bf16 bz = __builtin_bit_cast(__bf16, (unsigned short)0);
  for (int i = tid; i < 16 * H_; i += 512) {
    c0[i] = 0.f; c1[i] = 0.f; h0[i] = bz; h1[i] = bz;
  }
  __syncthreads();

  // -------- encoder --------
  for (int t = 0; t < T_IN; ++t) {
    const __bf16* xt = xb + (size_t)b0 * T_IN * F_ + (size_t)t * F_;
    gemm_gates(xt, T_IN * F_, eWih0, F_, F_,  h0, H_, eWhh0, H_, H_, eb0, gates, wave, lane);
    __syncthreads();
    lstm_cell(gates, c0, h0, tid);
    __syncthreads();
    gemm_gates(h0, H_, eWih1, H_, H_,  h1, H_, eWhh1, H_, H_, eb1, gates, wave, lane);
    __syncthreads();
    lstm_cell(gates, c1, h1, tid);
    __syncthreads();
  }

  // y0 = x[:, T_IN-1, :]
  for (int i = tid; i < 16 * F_; i += 512) {
    int m = i >> 5, f = i & 31;
    yb[i] = xb[((size_t)(b0 + m) * T_IN + (T_IN - 1)) * F_ + f];
  }
  __syncthreads();

  // -------- decoder --------
  for (int t = 0; t < T_OUT; ++t) {
    gemm_gates(yb, F_, dWih0, F_, F_,  h0, H_, dWhh0, H_, H_, db0, gates, wave, lane);
    __syncthreads();
    lstm_cell(gates, c0, h0, tid);
    __syncthreads();
    gemm_gates(h0, H_, dWih1, H_, H_,  h1, H_, dWhh1, H_, H_, db1, gates, wave, lane);
    __syncthreads();
    lstm_cell(gates, c1, h1, tid);
    __syncthreads();

    // y = h1 @ Wout^T + bout : 16x32, waves 0..1 (16 cols each)
    if (wave < 2) {
      const int nib = lane & 15, half = lane >> 4;
      const int n = wave * 16 + nib;
      float bb = dbout[n];
      v8f acc;
#pragma unroll
      for (int i = 0; i < 8; ++i) acc[i] = bb;
#pragma unroll 1
      for (int kt = 0; kt < H_; kt += 32) {
        v16bf a = load_a(h1, H_, kt, lane);
        v16bf b = load_b(dWout, H_, n, kt, lane);
        acc = __builtin_amdgcn_wmma_f32_16x16x32_bf16(
            false, a, false, b, (short)0, acc, false, false);
      }
#pragma unroll
      for (int r = 0; r < 8; ++r) {
        int m = r + half * 8;
        out[((size_t)(b0 + m) * T_OUT + t) * F_ + n] = acc[r];
        yb[m * F_ + n] = f2bf(acc[r]);
      }
    }
    __syncthreads();
  }
}

extern "C" void kernel_launch(void* const* d_in, const int* in_sizes, int n_in,
                              void* d_out, int out_size, void* d_ws, size_t ws_size,
                              hipStream_t stream) {
  (void)in_sizes; (void)n_in; (void)out_size; (void)ws_size;
  const float* x      = (const float*)d_in[0];
  const float* eWih0f = (const float*)d_in[1];
  const float* eWhh0f = (const float*)d_in[2];
  const float* eb0    = (const float*)d_in[3];
  const float* eWih1f = (const float*)d_in[4];
  const float* eWhh1f = (const float*)d_in[5];
  const float* eb1    = (const float*)d_in[6];
  const float* dWih0f = (const float*)d_in[7];
  const float* dWhh0f = (const float*)d_in[8];
  const float* db0    = (const float*)d_in[9];
  const float* dWih1f = (const float*)d_in[10];
  const float* dWhh1f = (const float*)d_in[11];
  const float* db1    = (const float*)d_in[12];
  const float* dWoutf = (const float*)d_in[13];
  const float* dbout  = (const float*)d_in[14];

  __bf16* ws = (__bf16*)d_ws;
  size_t off = 0;
  auto take = [&](size_t n) { __bf16* p = ws + off; off += n; return p; };
  __bf16* xb    = take((size_t)B_ * T_IN * F_);
  __bf16* eWih0 = take((size_t)G_ * F_);
  __bf16* eWhh0 = take((size_t)G_ * H_);
  __bf16* eWih1 = take((size_t)G_ * H_);
  __bf16* eWhh1 = take((size_t)G_ * H_);
  __bf16* dWih0 = take((size_t)G_ * F_);
  __bf16* dWhh0 = take((size_t)G_ * H_);
  __bf16* dWih1 = take((size_t)G_ * H_);
  __bf16* dWhh1 = take((size_t)G_ * H_);
  __bf16* dWout = take((size_t)F_ * H_);

  auto pack = [&](const float* s, __bf16* d, int n) {
    f32_to_bf16_k<<<(n + 255) / 256, 256, 0, stream>>>(s, d, n);
  };
  pack(x,      xb,    B_ * T_IN * F_);
  pack(eWih0f, eWih0, G_ * F_);
  pack(eWhh0f, eWhh0, G_ * H_);
  pack(eWih1f, eWih1, G_ * H_);
  pack(eWhh1f, eWhh1, G_ * H_);
  pack(dWih0f, dWih0, G_ * F_);
  pack(dWhh0f, dWhh0, G_ * H_);
  pack(dWih1f, dWih1, G_ * H_);
  pack(dWhh1f, dWhh1, G_ * H_);
  pack(dWoutf, dWout, F_ * H_);

  const size_t lds_bytes = 65536 + 32768 + 16384 + 1024; // 113 KB
  lstm_seq2seq_k<<<B_ / 16, 512, lds_bytes, stream>>>(
      xb, eWih0, eWhh0, eb0, eWih1, eWhh1, eb1,
      dWih0, dWhh0, db0, dWih1, dWhh1, db1,
      dWout, dbout, (float*)d_out);
}
